// SuperNodePool_53790170415290
// MI455X (gfx1250) — matrix operand order repaired and checked by reading, compiled, not run on hardware
//
#include <hip/hip_runtime.h>
#include <hip/hip_bf16.h>

#define HIDDEN 256
#define HEADS  8
#define DH     32
#define NGRAPH 512
#define SCALE  0.17677669529663687f   // 1/sqrt(32)

typedef __attribute__((ext_vector_type(16))) __bf16 v16bf;
typedef __attribute__((ext_vector_type(8)))  float  v8f;
typedef __attribute__((ext_vector_type(4)))  float  v4f;

// ---------- helpers ----------

// fp32 -> bf16 via native convert (gfx1250 has native BF16 VALU)
__device__ __forceinline__ __bf16 f2bf(float f) { return (__bf16)f; }

// order-preserving float<->uint encoding so atomicMax(u32) == float max
__device__ __forceinline__ unsigned fenc(float f) {
    unsigned u = __float_as_uint(f);
    return (u & 0x80000000u) ? ~u : (u | 0x80000000u);
}
__device__ __forceinline__ float fdec(unsigned e) {
    unsigned u = (e & 0x80000000u) ? (e & 0x7FFFFFFFu) : ~e;
    return __uint_as_float(u);
}

__device__ __forceinline__ v8f wmma_bf16(v16bf a, v16bf b, v8f c) {
    // v_wmma_f32_16x16x32_bf16
    return __builtin_amdgcn_wmma_f32_16x16x32_bf16(
        /*neg_a=*/false, a, /*neg_b=*/false, b,
        /*c_mod=*/(short)0, c, /*reuse_a=*/false, /*reuse_b=*/false);
}

// A-fragment (16x32 bf16): lane = M (%16); half=lane/16 selects K chunks
//   half 0: K = {0..15}; half 1: K = {16..31}   (contiguous, matches B)
__device__ __forceinline__ v16bf load_a_frag(const float* xrow, int kbase, int half) {
    const v4f* p0 = (const v4f*)(xrow + kbase + half * 16);
    const v4f* p1 = (const v4f*)(xrow + kbase + half * 16 + 8);
    v4f a0 = p0[0], a1 = p0[1], b0 = p1[0], b1 = p1[1];
    v16bf r;
    r[0]  = f2bf(a0.x); r[1]  = f2bf(a0.y); r[2]  = f2bf(a0.z); r[3]  = f2bf(a0.w);
    r[4]  = f2bf(a1.x); r[5]  = f2bf(a1.y); r[6]  = f2bf(a1.z); r[7]  = f2bf(a1.w);
    r[8]  = f2bf(b0.x); r[9]  = f2bf(b0.y); r[10] = f2bf(b0.z); r[11] = f2bf(b0.w);
    r[12] = f2bf(b1.x); r[13] = f2bf(b1.y); r[14] = f2bf(b1.z); r[15] = f2bf(b1.w);
    return r;
}

// ---------- kernel 0: zero accumulators, convert w_v to bf16 ----------
__global__ void k_init(const float* __restrict__ wv, __bf16* __restrict__ wvb,
                       unsigned* __restrict__ gmax, float* __restrict__ denom,
                       float* __restrict__ o_acc) {
    int i = blockIdx.x * blockDim.x + threadIdx.x;
    int stride = gridDim.x * blockDim.x;
    for (int j = i; j < HIDDEN * HIDDEN; j += stride) wvb[j] = f2bf(wv[j]);
    for (int j = i; j < NGRAPH * 16; j += stride) { gmax[j] = 0u; denom[j] = 0.f; }
    for (int j = i; j < NGRAPH * HEADS * 2 * DH; j += stride) o_acc[j] = 0.f;
}

// ---------- kernel 1: q = query@w_q^T + b_q ; build A_s^T (bf16) and c_s ----------
__global__ void k_prep_q(const float* __restrict__ query, const float* __restrict__ wq,
                         const float* __restrict__ bq, const float* __restrict__ wk,
                         const float* __restrict__ bk,
                         __bf16* __restrict__ asT, float* __restrict__ cs) {
    __shared__ float qs[2][HIDDEN];
    int i = threadIdx.x;  // 256 threads
    for (int t = 0; t < 2; t++) {
        float acc = bq[i];
        const float* w = wq + (size_t)i * HIDDEN;
        const float* qq = query + t * HIDDEN;
        for (int j = 0; j < HIDDEN; j++) acc += qq[j] * w[j];
        qs[t][i] = acc;
    }
    __syncthreads();
    // A_s^T[ht][j] = scale * sum_d w_k[h*32+d, j] * q[t][h*32+d]
    for (int ht = 0; ht < 16; ht++) {
        int h = ht >> 1, t = ht & 1;
        float acc = 0.f;
        for (int d = 0; d < DH; d++)
            acc += wk[(size_t)(h * DH + d) * HIDDEN + i] * qs[t][h * DH + d];
        asT[ht * HIDDEN + i] = f2bf(acc * SCALE);
    }
    if (i < 16) {
        int h = i >> 1, t = i & 1;
        float acc = 0.f;
        for (int d = 0; d < DH; d++) acc += bk[h * DH + d] * qs[t][h * DH + d];
        cs[i] = acc * SCALE;
    }
}

// ---------- kernel 2: s = x @ A_s + c_s (WMMA); store s; spill bf16 A-frags;
//                      atomic segment max ----------
__global__ void k_pass1(const float* __restrict__ x, const long long* __restrict__ batch,
                        const __bf16* __restrict__ asT, const float* __restrict__ cs,
                        float* __restrict__ s_ws, unsigned* __restrict__ gmax,
                        v16bf* __restrict__ frag_ws, int use_frags, int n_tiles) {
    int wave = threadIdx.x >> 5;
    int lane = threadIdx.x & 31;
    int tile = blockIdx.x * 8 + wave;
    if (tile >= n_tiles) return;
    int n0 = tile * 16;
    int lm = lane & 15, half = lane >> 4;

    const float* xrow = x + (size_t)(n0 + lm) * HIDDEN;
    v16bf a[8];
#pragma unroll
    for (int kk = 0; kk < 8; kk++) a[kk] = load_a_frag(xrow, kk * 32, half);

    if (use_frags) {
        // layout: frag_ws[(tile*8 + kk)*32 + lane]  (32B per lane, coalesced)
        v16bf* fp = frag_ws + (size_t)tile * 8 * 32 + lane;
#pragma unroll
        for (int kk = 0; kk < 8; kk++) fp[kk * 32] = a[kk];
    }

    v8f acc = {};
#pragma unroll
    for (int kk = 0; kk < 8; kk++) {
        v16bf b = *(const v16bf*)(asT + lm * HIDDEN + kk * 32 + half * 16);
        acc = wmma_bf16(a[kk], b, acc);
    }
    float csv = cs[lm];
    float sv[8];
    float mx = -3.4e38f;
#pragma unroll
    for (int r = 0; r < 8; r++) { sv[r] = acc[r] + csv; mx = fmaxf(mx, sv[r]); }

    // s layout: s_ws[tile*256 + ht*16 + M]  (lane: ht = lm, M = r + 8*half)
    float* sp = s_ws + (size_t)tile * 256 + lm * 16 + half * 8;
    v4f lo = {sv[0], sv[1], sv[2], sv[3]};
    v4f hi = {sv[4], sv[5], sv[6], sv[7]};
    *(v4f*)(sp) = lo;
    *(v4f*)(sp + 4) = hi;

    long long g0 = batch[n0];
    bool uniform = (batch[n0 + 15] == g0);
    if (uniform) {
        mx = fmaxf(mx, __shfl_xor(mx, 16));
        if (half == 0) atomicMax(&gmax[(int)g0 * 16 + lm], fenc(mx));
    } else {
#pragma unroll
        for (int r = 0; r < 8; r++) {
            long long g = batch[n0 + r + 8 * half];
            atomicMax(&gmax[(int)g * 16 + lm], fenc(sv[r]));
        }
    }
}

// ---------- kernel 3: p = exp(s - max); denom += p; o_acc += p * (x@w_v^T + b_v) ----------
__global__ void k_pass2(const float* __restrict__ x, const long long* __restrict__ batch,
                        const __bf16* __restrict__ wvb, const float* __restrict__ bv,
                        const float* __restrict__ s_ws, const unsigned* __restrict__ gmax,
                        float* __restrict__ denom, float* __restrict__ o_acc,
                        const v16bf* __restrict__ frag_ws, int use_frags, int n_tiles) {
    __shared__ float pbuf[8][256];
    int wave = threadIdx.x >> 5;
    int lane = threadIdx.x & 31;
    int tile = blockIdx.x * 8 + wave;
    bool active = (tile < n_tiles);
    int n0 = tile * 16;
    int lm = lane & 15, half = lane >> 4;
    long long g0 = 0;
    bool uniform = true;

    if (active) {
        g0 = batch[n0];
        uniform = (batch[n0 + 15] == g0);

        const float* sp = s_ws + (size_t)tile * 256 + lane * 8;
        v4f s0 = *(const v4f*)sp;
        v4f s1 = *(const v4f*)(sp + 4);
        float sv[8] = {s0.x, s0.y, s0.z, s0.w, s1.x, s1.y, s1.z, s1.w};
        int ht = lane >> 1;
        int mbase = (lane & 1) * 8;
        float p[8];
        if (uniform) {
            float mxv = fdec(gmax[(int)g0 * 16 + ht]);
#pragma unroll
            for (int j = 0; j < 8; j++) p[j] = __expf(sv[j] - mxv);
        } else {
#pragma unroll
            for (int j = 0; j < 8; j++) {
                long long g = batch[n0 + mbase + j];
                p[j] = __expf(sv[j] - fdec(gmax[(int)g * 16 + ht]));
            }
        }
#pragma unroll
        for (int j = 0; j < 8; j++) pbuf[wave][lane * 8 + j] = p[j];
    }
    __syncthreads();
    if (!active) return;

    // denom accumulation (unnormalized; divide in finalize)
    if (lane < 16) {
        if (uniform) {
            float s = 0.f;
#pragma unroll
            for (int m = 0; m < 16; m++) s += pbuf[wave][lane * 16 + m];
            atomicAdd(&denom[(int)g0 * 16 + lane], s);
        } else {
            int M = lane;
            long long g = batch[n0 + M];
            for (int ht2 = 0; ht2 < 16; ht2++)
                atomicAdd(&denom[(int)g * 16 + ht2], pbuf[wave][ht2 * 16 + M]);
        }
    }

    v16bf a[8];
    if (use_frags) {
        const v16bf* fp = frag_ws + (size_t)tile * 8 * 32 + lane;
#pragma unroll
        for (int kk = 0; kk < 8; kk++) a[kk] = fp[kk * 32];
    } else {
        const float* xrow = x + (size_t)(n0 + lm) * HIDDEN;
#pragma unroll
        for (int kk = 0; kk < 8; kk++) a[kk] = load_a_frag(xrow, kk * 32, half);
    }

    for (int h = 0; h < HEADS; h++) {
        float p0[8], p1[8];
#pragma unroll
        for (int r = 0; r < 8; r++) {
            int M = r + 8 * half;
            p0[r] = pbuf[wave][(2 * h + 0) * 16 + M];
            p1[r] = pbuf[wave][(2 * h + 1) * 16 + M];
        }
#pragma unroll
        for (int sub = 0; sub < 2; sub++) {
            int it = 2 * h + sub;
            int i0 = it * 16;
            v8f acc = {};
#pragma unroll
            for (int kk = 0; kk < 8; kk++) {
                v16bf b = *(const v16bf*)(wvb + (size_t)(i0 + lm) * HIDDEN + kk * 32 + half * 16);
                acc = wmma_bf16(a[kk], b, acc);
            }
            float bvv = bv[i0 + lm];
            int d = sub * 16 + lm;  // channel index within head
            if (uniform) {
                float c0 = 0.f, c1 = 0.f;
#pragma unroll
                for (int r = 0; r < 8; r++) {
                    float v = acc[r] + bvv;
                    c0 += p0[r] * v;
                    c1 += p1[r] * v;
                }
                c0 += __shfl_xor(c0, 16);
                c1 += __shfl_xor(c1, 16);
                if (half == 0) {
                    float* ob = o_acc + (((size_t)g0 * HEADS + h) * 2) * DH + d;
                    atomicAdd(ob, c0);
                    atomicAdd(ob + DH, c1);
                }
            } else {
#pragma unroll
                for (int r = 0; r < 8; r++) {
                    int M = r + 8 * half;
                    long long g = batch[n0 + M];
                    float v = acc[r] + bvv;
                    float* ob = o_acc + (((size_t)g * HEADS + h) * 2) * DH + d;
                    atomicAdd(ob, p0[r] * v);
                    atomicAdd(ob + DH, p1[r] * v);
                }
            }
        }
    }
}

// ---------- kernel 4: normalize, output projection, LayerNorm ----------
__global__ void k_final(const float* __restrict__ o_acc, const float* __restrict__ denom,
                        const float* __restrict__ wo, const float* __restrict__ bo,
                        const float* __restrict__ lg, const float* __restrict__ lb,
                        float* __restrict__ out) {
    __shared__ float u[HIDDEN];
    __shared__ float red[HIDDEN];
    int gt = blockIdx.x;
    int g = gt >> 1, t = gt & 1;
    int i = threadIdx.x;
    int h = i >> 5, d = i & 31;

    float dn = denom[g * 16 + h * 2 + t];
    u[i] = o_acc[(((size_t)g * HEADS + h) * 2 + t) * DH + d] / dn;
    __syncthreads();

    const v4f* wv4 = (const v4f*)(wo + (size_t)i * HIDDEN);
    const v4f* uv = (const v4f*)u;
    float y = bo[i];
#pragma unroll 4
    for (int j = 0; j < HIDDEN / 4; j++) {
        v4f a = uv[j], b = wv4[j];
        y += a.x * b.x + a.y * b.y + a.z * b.z + a.w * b.w;
    }

    red[i] = y;
    __syncthreads();
    for (int s = 128; s > 0; s >>= 1) { if (i < s) red[i] += red[i + s]; __syncthreads(); }
    float mu = red[0] * (1.f / HIDDEN);
    __syncthreads();
    float yd = y - mu;
    red[i] = yd * yd;
    __syncthreads();
    for (int s = 128; s > 0; s >>= 1) { if (i < s) red[i] += red[i + s]; __syncthreads(); }
    float var = red[0] * (1.f / HIDDEN);
    float r = rsqrtf(var + 1e-5f);
    out[(size_t)gt * HIDDEN + i] = yd * r * lg[i] + lb[i];
}

// ---------- launcher ----------
extern "C" void kernel_launch(void* const* d_in, const int* in_sizes, int n_in,
                              void* d_out, int out_size, void* d_ws, size_t ws_size,
                              hipStream_t stream) {
    const float*     x     = (const float*)d_in[0];
    const long long* batch = (const long long*)d_in[1];
    // d_in[2] = num_graphs (scalar, == 512 per setup)
    const float* query = (const float*)d_in[3];
    const float* w_q   = (const float*)d_in[4];
    const float* b_q   = (const float*)d_in[5];
    const float* w_k   = (const float*)d_in[6];
    const float* b_k   = (const float*)d_in[7];
    const float* w_v   = (const float*)d_in[8];
    const float* b_v   = (const float*)d_in[9];
    const float* w_o   = (const float*)d_in[10];
    const float* b_o   = (const float*)d_in[11];
    const float* ln_g  = (const float*)d_in[12];
    const float* ln_b  = (const float*)d_in[13];

    const int N = in_sizes[0] / HIDDEN;          // 131072
    const int n_tiles = (N + 15) / 16;           // 8192
    const int blocks_p = (n_tiles + 7) / 8;      // 1024 (8 waves / block)

    // workspace carve-up
    char* base = (char*)d_ws;
    size_t off = 0;
    float*    cs    = (float*)(base + off);    off += 256;
    __bf16*   asT   = (__bf16*)(base + off);   off += (size_t)16 * HIDDEN * 2;          // 8 KB
    __bf16*   wvb   = (__bf16*)(base + off);   off += (size_t)HIDDEN * HIDDEN * 2;      // 128 KB
    unsigned* gmax  = (unsigned*)(base + off); off += (size_t)NGRAPH * 16 * 4;          // 32 KB
    float*    denom = (float*)(base + off);    off += (size_t)NGRAPH * 16 * 4;          // 32 KB
    float*    o_acc = (float*)(base + off);    off += (size_t)NGRAPH * HEADS * 2 * DH * 4; // 1 MB
    float*    s_ws  = (float*)(base + off);    off += (size_t)n_tiles * 256 * 4;        // 8 MB
    // optional bf16 A-fragment spill: n_tiles * 8 frags * 32 lanes * 32 B  (~67 MB)
    size_t frag_bytes = (size_t)n_tiles * 8 * 32 * sizeof(v16bf);
    v16bf* frag_ws = (v16bf*)(base + off);
    int use_frags = (ws_size >= off + frag_bytes) ? 1 : 0;

    k_init<<<256, 256, 0, stream>>>(w_v, wvb, gmax, denom, o_acc);
    k_prep_q<<<1, 256, 0, stream>>>(query, w_q, b_q, w_k, b_k, asT, cs);
    k_pass1<<<blocks_p, 256, 0, stream>>>(x, batch, asT, cs, s_ws, gmax,
                                          frag_ws, use_frags, n_tiles);
    k_pass2<<<blocks_p, 256, 0, stream>>>(x, batch, wvb, b_v, s_ws, gmax, denom, o_acc,
                                          frag_ws, use_frags, n_tiles);
    k_final<<<NGRAPH * 2, 256, 0, stream>>>(o_acc, denom, w_o, b_o, ln_g, ln_b, (float*)d_out);
}
